// OuterProduct_89438398972452
// MI455X (gfx1250) — matrix operand order repaired and checked by reading, not compiled, not run on hardware
//
#include <hip/hip_runtime.h>
#include <hip/hip_bf16.h>

// Problem constants (match the reference)
#define DIMX   1024
#define HEADS  16
#define DHEAD  64
#define HIDX   1024
#define NBATCH 2
#define TSEQ   1024
#define MROWS  (NBATCH * TSEQ)
#define CHUNK  16

typedef __attribute__((ext_vector_type(16))) __bf16 v16bf;
typedef __attribute__((ext_vector_type(8)))  __bf16 v8bf;
typedef __attribute__((ext_vector_type(8)))  float  v8f;
typedef __attribute__((ext_vector_type(4)))  int    v4i;

#define WMMA(a, b, c) \
  __builtin_amdgcn_wmma_f32_16x16x32_bf16(false, (a), false, (b), (short)0, (c), false, false)

// ---- CDNA5 async global->LDS copy (ASYNCcnt-tracked, bypasses VGPRs) -------
static __device__ inline void async_copy_b128(const float* gsrc, float* ldst) {
#if __has_builtin(__builtin_amdgcn_global_load_async_to_lds_b128)
  __builtin_amdgcn_global_load_async_to_lds_b128(
      (__attribute__((address_space(1))) v4i*)gsrc,
      (__attribute__((address_space(3))) v4i*)ldst,
      /*offset=*/0, /*cpol=*/0);
#else
  asm volatile("global_load_async_to_lds_b128 %0, %1, off"
               :: "v"((unsigned)(unsigned long long)(void*)ldst),
                  "v"(gsrc)
               : "memory");
#endif
}

#if __has_builtin(__builtin_amdgcn_s_wait_asynccnt)
#define WAIT_ASYNC(n) __builtin_amdgcn_s_wait_asynccnt(n)
#else
#define WAIT_ASYNC(n) asm volatile("s_wait_asynccnt %0" :: "i"(n) : "memory")
#endif

// ---- WMMA fragment builders from fp32 LDS tiles (convert at load time) -----
// Layouts per ISA 7.12.2 (wave32):
//  A (16-bit, 16x32): lanes 0-15 -> row=lane, K {0..7}U{16..23}; lanes 16-31 ->
//  row=lane-16, K {8..15}U{24..31}.
//  B (16-bit, 32x16): lanes 0-15 -> col=lane, K {0..7}U{16..23}; lanes 16-31 -> K {8..15}U{24..31}.
static __device__ inline v8bf cvt8(float4 x, float4 y) {
  v8bf r;
  r[0] = (__bf16)x.x; r[1] = (__bf16)x.y; r[2] = (__bf16)x.z; r[3] = (__bf16)x.w;
  r[4] = (__bf16)y.x; r[5] = (__bf16)y.y; r[6] = (__bf16)y.z; r[7] = (__bf16)y.w;
  return r;
}
static __device__ inline v16bf frag_a_f32(const float* buf, int row, int ld, int kc) {
  const int lane = threadIdx.x & 31;
  const int r = row + (lane & 15);
  const int k0 = kc + ((lane & 16) ? 8 : 0);
  const float4* p  = (const float4*)(buf + r * ld + k0);
  const float4* p2 = (const float4*)(buf + r * ld + k0 + 16);
  v8bf lo = cvt8(p[0], p[1]);
  v8bf hi = cvt8(p2[0], p2[1]);
  return __builtin_shufflevector(lo, hi, 0,1,2,3,4,5,6,7,8,9,10,11,12,13,14,15);
}
static __device__ inline v16bf frag_b_f32(const float* buf, int row, int ld, int kc) {
  const int lane = threadIdx.x & 31;
  const int r = row + (lane & 15);
  const int k0 = kc + ((lane & 16) ? 8 : 0);
  const float4* p = (const float4*)(buf + r * ld + k0);
  v8bf lo = cvt8(p[0], p[1]);
  v8bf hi = cvt8(p[4], p[5]);
  return __builtin_shufflevector(lo, hi, 0,1,2,3,4,5,6,7,8,9,10,11,12,13,14,15);
}

// ---- bf16-LDS fragment loaders (used by the recurrence kernel) -------------
static __device__ inline v16bf frag_a(const __bf16* buf, int row, int ld, int kc) {
  const int lane = threadIdx.x & 31;
  const int r = row + (lane & 15);
  const int k0 = kc + ((lane & 16) ? 8 : 0);
  v8bf lo = *(const v8bf*)(buf + r * ld + k0);
  v8bf hi = *(const v8bf*)(buf + r * ld + k0 + 16);
  return __builtin_shufflevector(lo, hi, 0,1,2,3,4,5,6,7,8,9,10,11,12,13,14,15);
}
static __device__ inline v16bf frag_a_k16(const __bf16* buf, int row, int ld) {
  const int lane = threadIdx.x & 31;
  const int r = row + (lane & 15);
  const int k0 = (lane & 16) ? 8 : 0;
  v8bf lo = *(const v8bf*)(buf + r * ld + k0);
  v8bf z = {};
  return __builtin_shufflevector(lo, z, 0,1,2,3,4,5,6,7,8,9,10,11,12,13,14,15);
}
static __device__ inline v16bf frag_b(const __bf16* buf, int row, int ld, int kc) {
  const int lane = threadIdx.x & 31;
  const int r = row + (lane & 15);
  const int k0 = kc + ((lane & 16) ? 8 : 0);
  v8bf lo = *(const v8bf*)(buf + r * ld + k0);
  v8bf hi = *(const v8bf*)(buf + r * ld + k0 + 16);
  return __builtin_shufflevector(lo, hi, 0,1,2,3,4,5,6,7,8,9,10,11,12,13,14,15);
}
static __device__ inline v16bf frag_b_k16(const __bf16* buf, int row, int ld) {
  const int lane = threadIdx.x & 31;
  const int r = row + (lane & 15);
  v8bf lo = *(const v8bf*)(buf + r * ld + ((lane & 16) ? 8 : 0));
  v8bf z = {};
  return __builtin_shufflevector(lo, z, 0,1,2,3,4,5,6,7,8,9,10,11,12,13,14,15);
}

// ---- Generic fused GEMM: O = act(X * W^T + bias) ---------------------------
// X: MxK fp32 row-major, W: NnxK fp32 row-major, O: MxNn fp32.
// 256 thr / 8 waves per 64x64 tile; fp32 tiles staged to LDS via async copies,
// double-buffered; bf16 conversion happens on the LDS->VGPR fragment path.
__global__ __launch_bounds__(256)
void gemm_bf16_act(const float* __restrict__ X, const float* __restrict__ W,
                   const float* __restrict__ bias, float* __restrict__ O,
                   int M, int Nn, int K, int do_sigmoid)
{
  __shared__ __attribute__((aligned(32))) float a_s[2][64 * 32];
  __shared__ __attribute__((aligned(32))) float b_s[2][64 * 32];
  const int tid  = threadIdx.x;
  const int lane = tid & 31;
  const int wave = tid >> 5;
  const int ntiles = Nn >> 6;
  const int bm = (blockIdx.x / ntiles) << 6;
  const int bn = (blockIdx.x % ntiles) << 6;
  const int mT  = wave >> 1;          // 0..3 (16-row tile)
  const int nT0 = (wave & 1) << 1;    // 0 or 2 (two 16-col tiles)
  v8f acc0 = {}; v8f acc1 = {};

  // thread -> two float4 chunks of each 64x32 tile (512 chunks, 256 threads)
  const int c0 = tid, c1 = tid + 256;
  const int r0 = c0 >> 3, o0 = (c0 & 7) << 2;
  const int r1 = c1 >> 3, o1 = (c1 & 7) << 2;

#define STAGE(kc, buf)                                                          \
  do {                                                                          \
    async_copy_b128(X + (size_t)(bm + r0) * K + (kc) + o0, a_s[buf] + r0 * 32 + o0); \
    async_copy_b128(X + (size_t)(bm + r1) * K + (kc) + o1, a_s[buf] + r1 * 32 + o1); \
    async_copy_b128(W + (size_t)(bn + r0) * K + (kc) + o0, b_s[buf] + r0 * 32 + o0); \
    async_copy_b128(W + (size_t)(bn + r1) * K + (kc) + o1, b_s[buf] + r1 * 32 + o1); \
  } while (0)

  const int nsteps = K >> 5;
  STAGE(0, 0);
  STAGE(32, 1);

  for (int i = 0; i < nsteps; ++i) {
    if (i + 1 < nsteps) { WAIT_ASYNC(4); } else { WAIT_ASYNC(0); }  // stage i landed
    __syncthreads();
    const float* ab = a_s[i & 1];
    const float* bb = b_s[i & 1];
    v16bf a  = frag_a_f32(ab, mT << 4, 32, 0);
    v16bf b0 = frag_b_f32(bb, nT0 << 4, 32, 0);
    v16bf b1 = frag_b_f32(bb, (nT0 + 1) << 4, 32, 0);
    acc0 = WMMA(a, b0, acc0);
    acc1 = WMMA(a, b1, acc1);
    __syncthreads();                    // everyone done reading buf[i&1]
    if (i + 2 < nsteps) STAGE((i + 2) << 5, i & 1);
  }
#undef STAGE

  const int nl = lane & 15;
  const int mh = (lane & 16) ? 8 : 0;   // C layout: VGPR r -> M=r (lanes 0-15) / r+8
#pragma unroll
  for (int r = 0; r < 8; ++r) {
    int m  = bm + (mT << 4) + r + mh;
    int n0 = bn + (nT0 << 4) + nl;
    int n1 = bn + ((nT0 + 1) << 4) + nl;
    float v0 = acc0[r] + (bias ? bias[n0] : 0.f);
    float v1 = acc1[r] + (bias ? bias[n1] : 0.f);
    if (do_sigmoid) {
      v0 = 1.f / (1.f + __expf(-v0));
      v1 = 1.f / (1.f + __expf(-v1));
    }
    O[(size_t)m * Nn + n0] = v0;
    O[(size_t)m * Nn + n1] = v1;
  }
}

// ---- Chunked gated recurrence (linear-attention chunk form) ----------------
// One workgroup per (batch, head); 4 waves. Per chunk of 16 timesteps:
//   A[t,d] = prod_{r<=t} k[r,d]      (inclusive cumprod, fp32)
//   Q'[t,d] = q[t,d]*A[t,d];  W[s,d] = (1-k[s,d])/A[s,d]
//   P = tril(Q' W^T)                        (16x16x64 -> 2 WMMAs)
//   h = P V + Q' S0                         (per 16-col e-tile: 3 WMMAs)
//   S = diag(A_last) S + W2^T V,  W2[s,d] = A_last[d]/A[s,d]*(1-k[s,d])
__global__ __launch_bounds__(128)
void chunk_recurrence(const float* __restrict__ kk, const float* __restrict__ vv,
                      const float* __restrict__ qq, float* __restrict__ hh)
{
  __shared__ __attribute__((aligned(32))) float  S[DHEAD * DHEAD];     // [d][e] fp32
  __shared__ __attribute__((aligned(32))) __bf16 Sb[DHEAD * DHEAD];    // [e][d] bf16 snapshot
  __shared__ __attribute__((aligned(32))) float  k_s[CHUNK * DHEAD];   // [t][d]
  __shared__ __attribute__((aligned(32))) float  q_s[CHUNK * DHEAD];
  __shared__ __attribute__((aligned(32))) float  A_s[CHUNK * DHEAD];
  __shared__ float A_last[DHEAD];
  __shared__ __attribute__((aligned(32))) __bf16 vT [DHEAD * CHUNK];   // [e][s]
  __shared__ __attribute__((aligned(32))) __bf16 qp [CHUNK * DHEAD];   // [t][d]
  __shared__ __attribute__((aligned(32))) __bf16 wB [CHUNK * DHEAD];   // [s][d]
  __shared__ __attribute__((aligned(32))) __bf16 w2T[DHEAD * CHUNK];   // [d][s]
  __shared__ __attribute__((aligned(32))) __bf16 p_s[4][CHUNK * CHUNK];// per-wave tril(P)

  const int tid  = threadIdx.x;
  const int lane = tid & 31;
  const int wave = tid >> 5;
  const int nb   = blockIdx.x >> 4;
  const int head = blockIdx.x & 15;
  const size_t base = ((size_t)nb * TSEQ) * HIDX + (size_t)head * DHEAD;

  for (int i = tid; i < DHEAD * DHEAD; i += 128) S[i] = 0.f;
  __syncthreads();

  for (int t0 = 0; t0 < TSEQ; t0 += CHUNK) {
    // Prefetch next chunk's rows while we work on this one
    if (t0 + CHUNK < TSEQ && tid < 48) {
      int t = tid & 15, w = tid >> 4;
      size_t g = base + (size_t)(t0 + CHUNK + t) * HIDX;
      const float* p = (w == 0) ? kk : (w == 1) ? qq : vv;
      __builtin_prefetch(p + g, 0, 1);
      __builtin_prefetch(p + g + 32, 0, 1);
    }
    // Stage chunk inputs + snapshot state (bf16, transposed for B-frag reads)
    for (int i = tid; i < CHUNK * DHEAD; i += 128) {
      int t = i >> 6, d = i & 63;
      size_t g = base + (size_t)(t0 + t) * HIDX + d;
      float kv = kk[g];
      k_s[i] = kv;
      q_s[i] = qq[g];
      vT[d * CHUNK + t] = (__bf16)vv[g];   // here d plays the role of e
    }
    for (int i = tid; i < DHEAD * DHEAD; i += 128) {
      int d = i >> 6, e = i & 63;
      Sb[e * DHEAD + d] = (__bf16)S[i];
    }
    __syncthreads();

    // Inclusive cumprod of decays along t (64 independent d-lanes)
    if (tid < DHEAD) {
      float a = 1.f;
#pragma unroll
      for (int t = 0; t < CHUNK; ++t) {
        a *= k_s[t * DHEAD + tid];
        A_s[t * DHEAD + tid] = a;
      }
      A_last[tid] = a;
    }
    __syncthreads();

    // Derived bf16 operands
    for (int i = tid; i < CHUNK * DHEAD; i += 128) {
      int t = i >> 6, d = i & 63;
      float kv  = k_s[i];
      float Aa  = A_s[i];
      float inv = __builtin_amdgcn_rcpf(Aa);
      qp[i] = (__bf16)(q_s[i] * Aa);
      wB[i] = (__bf16)((1.f - kv) * inv);
      w2T[d * CHUNK + t] = (__bf16)(A_last[d] * inv * (1.f - kv));
    }
    __syncthreads();

    // P = Q' W^T (16x16, K=64), computed per wave, causal-masked into LDS
    v8f pc = {};
    pc = WMMA(frag_a(qp, 0, 64, 0),  frag_b(wB, 0, 64, 0),  pc);
    pc = WMMA(frag_a(qp, 0, 64, 32), frag_b(wB, 0, 64, 32), pc);
    {
      const int sC = lane & 15;
      const int mh = (lane & 16) ? 8 : 0;
#pragma unroll
      for (int r = 0; r < 8; ++r) {
        int t = r + mh;
        p_s[wave][t * CHUNK + sC] = (__bf16)((sC <= t) ? pc[r] : 0.f);
      }
    }

    // h tile (e-tile = wave): h = P V + Q' S0
    v8f hc = {};
    hc = WMMA(frag_a_k16(p_s[wave], 0, CHUNK), frag_b_k16(vT, wave << 4, CHUNK), hc);
    hc = WMMA(frag_a(qp, 0, 64, 0),  frag_b(Sb, wave << 4, 64, 0),  hc);
    hc = WMMA(frag_a(qp, 0, 64, 32), frag_b(Sb, wave << 4, 64, 32), hc);
    {
      const int e  = (wave << 4) + (lane & 15);
      const int mh = (lane & 16) ? 8 : 0;
#pragma unroll
      for (int r = 0; r < 8; ++r) {
        int t = t0 + r + mh;
        hh[((size_t)nb * TSEQ + t) * HIDX + (size_t)head * DHEAD + e] = hc[r];
      }
    }

    // State update (d-tile = wave): S = diag(A_last) S + W2^T V
    v16bf wa = frag_a_k16(w2T, wave << 4, CHUNK);
#pragma unroll
    for (int et = 0; et < 4; ++et) {
      v8f sc = {};
      sc = WMMA(wa, frag_b_k16(vT, et << 4, CHUNK), sc);
      const int e  = (et << 4) + (lane & 15);
      const int mh = (lane & 16) ? 8 : 0;
#pragma unroll
      for (int r = 0; r < 8; ++r) {
        int d = (wave << 4) + r + mh;
        S[d * DHEAD + e] = A_last[d] * S[d * DHEAD + e] + sc[r];
      }
    }
    __syncthreads();
  }
}

// ---- Host-side launch ------------------------------------------------------
extern "C" void kernel_launch(void* const* d_in, const int* in_sizes, int n_in,
                              void* d_out, int out_size, void* d_ws, size_t ws_size,
                              hipStream_t stream)
{
  (void)in_sizes; (void)n_in; (void)out_size; (void)ws_size;
  const float* x     = (const float*)d_in[0];
  const float* key_w = (const float*)d_in[1];
  const float* key_b = (const float*)d_in[2];
  const float* vq_w  = (const float*)d_in[3];
  const float* out_w = (const float*)d_in[4];
  const float* out_b = (const float*)d_in[5];
  float* out = (float*)d_out;

  // Workspace: k, v, q, h planes (each MROWS*HIDX fp32) => 32 MB total.
  const size_t plane = (size_t)MROWS * HIDX;
  float* k_ws = (float*)d_ws;
  float* v_ws = k_ws + plane;
  float* q_ws = v_ws + plane;
  float* h_ws = q_ws + plane;

  const int g_in  = (MROWS / 64) * (HIDX / 64);   // 512 blocks
  const int g_out = (MROWS / 64) * (DIMX / 64);   // 512 blocks

  gemm_bf16_act<<<g_in, 256, 0, stream>>>(x, key_w, key_b, k_ws, MROWS, HIDX, DIMX, 1);
  gemm_bf16_act<<<g_in, 256, 0, stream>>>(x, vq_w, nullptr, v_ws, MROWS, HIDX, DIMX, 0);
  gemm_bf16_act<<<g_in, 256, 0, stream>>>(x, vq_w + (size_t)HIDX * DIMX, nullptr, q_ws,
                                          MROWS, HIDX, DIMX, 0);
  chunk_recurrence<<<NBATCH * HEADS, 128, 0, stream>>>(k_ws, v_ws, q_ws, h_ws);
  gemm_bf16_act<<<g_out, 256, 0, stream>>>(h_ws, out_w, out_b, out, MROWS, DIMX, HIDX, 0);
}
